// SeqDecoder_3813930959097
// MI455X (gfx1250) — compile-verified, matrix operand order checked by reading, not run on hardware
//
#include <hip/hip_runtime.h>

#define Bb 256
#define Ss 512
#define Ee 512
#define Uu 512
#define Oo 256
#define Tt 16
#define U4 (4*Uu)

typedef __attribute__((ext_vector_type(16))) __bf16 v16bf;
typedef __attribute__((ext_vector_type(8)))  __bf16 v8bf;
typedef __attribute__((ext_vector_type(4)))  __bf16 v4bf;
typedef __attribute__((ext_vector_type(8)))  float  v8f;
typedef int v4i_ld __attribute__((vector_size(16)));   // matches builtin param type

#if defined(__has_builtin)
#if __has_builtin(__builtin_amdgcn_global_load_async_to_lds_b128)
#define HAVE_ASYNC_LDS 1
#endif
#endif

__device__ __forceinline__ __bf16 f2bf(float f) {
    unsigned int u = __builtin_bit_cast(unsigned int, f);
    unsigned int r = (u + 0x7FFFu + ((u >> 16) & 1u)) >> 16;   // RNE
    unsigned short s = (unsigned short)r;
    return __builtin_bit_cast(__bf16, s);
}
__device__ __forceinline__ float bflo(unsigned int u) {
    return __builtin_bit_cast(float, u << 16);
}
__device__ __forceinline__ float bfhi(unsigned int u) {
    return __builtin_bit_cast(float, u & 0xFFFF0000u);
}

__device__ __forceinline__ void async_wait_le2() {
#ifdef HAVE_ASYNC_LDS
#if __has_builtin(__builtin_amdgcn_s_wait_asynccnt)
    __builtin_amdgcn_s_wait_asynccnt(2);
#else
    asm volatile("s_wait_asynccnt 0x2" ::: "memory");
#endif
#endif
}
__device__ __forceinline__ void async_wait_le0() {
#ifdef HAVE_ASYNC_LDS
#if __has_builtin(__builtin_amdgcn_s_wait_asynccnt)
    __builtin_amdgcn_s_wait_asynccnt(0);
#else
    asm volatile("s_wait_asynccnt 0x0" ::: "memory");
#endif
#endif
}

// ---- elementwise f32 -> bf16 (4 per thread) ----
__global__ __launch_bounds__(256) void cvt_bf16_kernel(
    const float* __restrict__ in, __bf16* __restrict__ out, int n4)
{
    int i = blockIdx.x * blockDim.x + threadIdx.x;
    if (i >= n4) return;
    float4 v = ((const float4*)in)[i];
    v4bf o; o[0] = f2bf(v.x); o[1] = f2bf(v.y); o[2] = f2bf(v.z); o[3] = f2bf(v.w);
    ((v4bf*)out)[i] = o;
}

// ---- [mat][K,N] f32 -> [mat][N,K] bf16, LDS 32x32 tile transpose ----
__global__ __launch_bounds__(256) void transpose_bf16_kernel(
    const float* __restrict__ in, __bf16* __restrict__ out, int K, int N)
{
    __shared__ float tile[32][33];
    const int mat = blockIdx.z;
    const int k0 = blockIdx.y * 32;
    const int n0 = blockIdx.x * 32;
    const float* src = in  + (size_t)mat * K * N;
    __bf16*      dst = out + (size_t)mat * K * N;
    const int tx = threadIdx.x & 31, ty = threadIdx.x >> 5;
#pragma unroll
    for (int i = 0; i < 32; i += 8)
        tile[ty + i][tx] = src[(size_t)(k0 + ty + i) * N + (n0 + tx)];
    __syncthreads();
#pragma unroll
    for (int i = 0; i < 32; i += 8)
        dst[(size_t)(n0 + ty + i) * K + (k0 + tx)] = f2bf(tile[tx][ty + i]);
}

// ---- block-cooperative GEMM ----
// C[M,N] (+)= A[M,K](bf16) * Bt[N,K](bf16)^T + bias ; out f32 or bf16.
// Block = 8 waves = 128 rows x 64 cols. The shared B-slab (64 cols x 64 k)
// is double-buffered in LDS via async global->LDS copies (ASYNCcnt).
#define KC 64
#define PSTR 72   // padded row stride in bf16 elements (144 B): conflict-free b128 reads

__global__ __launch_bounds__(256) void gemm_bf16_wmma(
    const __bf16* __restrict__ A, const __bf16* __restrict__ Bt,
    const float* __restrict__ bias,
    float* __restrict__ Cf, __bf16* __restrict__ Cb,
    int M, int N, int K, int ldc, int accumulate, int outBf)
{
    __shared__ __bf16 sB[2][64][PSTR];

    const int tid  = threadIdx.x;
    const int lane = tid & 31;
    const int wid  = tid >> 5;
    const int nTiles = N >> 6;
    const int bm = blockIdx.x / nTiles;
    const int bn = blockIdx.x % nTiles;
    const int m0 = bm * 128 + wid * 16;
    const int n0 = bn * 64;
    (void)M;

    v8f acc[4];
#pragma unroll
    for (int j = 0; j < 4; ++j) acc[j] = (v8f){0.f,0.f,0.f,0.f,0.f,0.f,0.f,0.f};

    const int khalf = (lane >> 4) << 3;                 // 0 or 8
    const int ncol  = lane & 15;
    const __bf16* pa = A + (size_t)(m0 + (lane & 15)) * K + khalf;

    // B-slab copy mapping: 2 rounds x 256 lanes x 16B = 8 KB
    const int row0 = tid >> 3;                          // 0..31 (round 0), +32 (round 1)
    const int cu   = tid & 7;                           // 16B unit within 128B row

    auto issue_slab = [&](int buf, int kc) {
        const __bf16* g0 = Bt + (size_t)(n0 + row0) * K + kc + cu * 8;
        const __bf16* g1 = Bt + (size_t)(n0 + row0 + 32) * K + kc + cu * 8;
        __bf16* l0 = &sB[buf][row0][cu * 8];
        __bf16* l1 = &sB[buf][row0 + 32][cu * 8];
#ifdef HAVE_ASYNC_LDS
        __builtin_amdgcn_global_load_async_to_lds_b128((v4i_ld*)g0, (v4i_ld*)l0, 0, 0);
        __builtin_amdgcn_global_load_async_to_lds_b128((v4i_ld*)g1, (v4i_ld*)l1, 0, 0);
#else
        *(uint4*)l0 = *(const uint4*)g0;
        *(uint4*)l1 = *(const uint4*)g1;
#endif
    };

    const int nChunks = K / KC;
    issue_slab(0, 0);
    for (int ch = 0; ch < nChunks; ++ch) {
        const int cur = ch & 1;
        const bool hasNext = (ch + 1) < nChunks;
        if (hasNext) issue_slab(cur ^ 1, (ch + 1) * KC);
        if (hasNext) async_wait_le2(); else async_wait_le0();
        __syncthreads();                                // slab `cur` visible to all

#pragma unroll
        for (int kk = 0; kk < 2; ++kk) {                // two K=32 sub-steps
            const int k0 = ch * KC + kk * 32;
            v8bf alo = *(const v8bf*)(pa + k0);
            v8bf ahi = *(const v8bf*)(pa + k0 + 16);
            v16bf af = __builtin_shufflevector(alo, ahi,
                0,1,2,3,4,5,6,7,8,9,10,11,12,13,14,15);
#pragma unroll
            for (int j = 0; j < 4; ++j) {
                const __bf16* pb = &sB[cur][j * 16 + ncol][khalf + kk * 32];
                v8bf blo = *(const v8bf*)(pb);
                v8bf bhi = *(const v8bf*)(pb + 16);
                v16bf bfrag = __builtin_shufflevector(blo, bhi,
                    0,1,2,3,4,5,6,7,8,9,10,11,12,13,14,15);
                acc[j] = __builtin_amdgcn_wmma_f32_16x16x32_bf16(
                    false, af, false, bfrag, (short)0, acc[j], false, false);
            }
        }
        __syncthreads();                                // done reading before refill
    }

    const int rbase = m0 + ((lane >> 4) << 3);
#pragma unroll
    for (int j = 0; j < 4; ++j) {
        const int col = n0 + j * 16 + ncol;
        const float bv = bias ? bias[col] : 0.f;
#pragma unroll
        for (int r = 0; r < 8; ++r) {
            const size_t idx = (size_t)(rbase + r) * ldc + col;
            float v = acc[j][r] + bv;
            if (accumulate) v += Cf[idx];
            if (outBf) Cb[idx] = f2bf(v); else Cf[idx] = v;
        }
    }
}

// ---- fused attention: one batch row per block (512 threads = 16 waves) ----
__global__ __launch_bounds__(512) void attention_kernel(
    const __bf16* __restrict__ ep, const float* __restrict__ q,
    const float* __restrict__ Vw, const float* __restrict__ Vb,
    const __bf16* __restrict__ enc, float* __restrict__ ctx,
    __bf16* __restrict__ ctx_bf, float* __restrict__ probs, int tstep)
{
    __shared__ float s_vw[Uu];
    __shared__ float s_q[Uu];
    __shared__ float s_sc[Ss];
    __shared__ float s_red[16];
    __shared__ float s_ctx[2][Ee];

    const int b    = blockIdx.x;
    const int tid  = threadIdx.x;
    const int lane = tid & 31;
    const int wid  = tid >> 5;

    s_vw[tid] = Vw[tid];
    s_q[tid]  = q[(size_t)b * Uu + tid];
    __syncthreads();

    const float vb = Vb[0];
    for (int j = 0; j < 32; ++j) {
        const int s = wid * 32 + j;
        const unsigned int* eprow =
            (const unsigned int*)(ep + ((size_t)b * Ss + s) * Uu);
        float acc = 0.f;
#pragma unroll
        for (int it = 0; it < 2; ++it) {
            const int e = it * 256 + lane * 8;
            uint4 u = *(const uint4*)(eprow + (e >> 1));
            acc += tanhf(bflo(u.x) + s_q[e + 0]) * s_vw[e + 0];
            acc += tanhf(bfhi(u.x) + s_q[e + 1]) * s_vw[e + 1];
            acc += tanhf(bflo(u.y) + s_q[e + 2]) * s_vw[e + 2];
            acc += tanhf(bfhi(u.y) + s_q[e + 3]) * s_vw[e + 3];
            acc += tanhf(bflo(u.z) + s_q[e + 4]) * s_vw[e + 4];
            acc += tanhf(bfhi(u.z) + s_q[e + 5]) * s_vw[e + 5];
            acc += tanhf(bflo(u.w) + s_q[e + 6]) * s_vw[e + 6];
            acc += tanhf(bfhi(u.w) + s_q[e + 7]) * s_vw[e + 7];
        }
#pragma unroll
        for (int off = 16; off > 0; off >>= 1) acc += __shfl_xor(acc, off, 32);
        if (lane == 0) s_sc[s] = acc + vb;
    }
    __syncthreads();

    const float v = s_sc[tid];
    float m = v;
#pragma unroll
    for (int off = 16; off > 0; off >>= 1) m = fmaxf(m, __shfl_xor(m, off, 32));
    if (lane == 0) s_red[wid] = m;
    __syncthreads();
    if (wid == 0) {
        float mm = s_red[lane & 15];
#pragma unroll
        for (int off = 8; off > 0; off >>= 1) mm = fmaxf(mm, __shfl_xor(mm, off, 32));
        if (lane == 0) s_red[0] = mm;
    }
    __syncthreads();
    const float gmax = s_red[0];
    const float ev = __expf(v - gmax);
    float ssum = ev;
#pragma unroll
    for (int off = 16; off > 0; off >>= 1) ssum += __shfl_xor(ssum, off, 32);
    __syncthreads();
    if (lane == 0) s_red[wid] = ssum;
    __syncthreads();
    if (wid == 0) {
        float t = s_red[lane & 15];
#pragma unroll
        for (int off = 8; off > 0; off >>= 1) t += __shfl_xor(t, off, 32);
        if (lane == 0) s_red[0] = t;
    }
    __syncthreads();
    const float aw = ev * (1.0f / s_red[0]);
    s_sc[tid] = aw;
    probs[((size_t)b * Tt + tstep) * Ss + tid] = aw;
    __syncthreads();

    {
        const unsigned int* eb =
            (const unsigned int*)(enc + (size_t)b * Ss * Ee) + (tid & 255);
        const int sbeg = (tid >> 8) << 8;
        float a0 = 0.f, a1 = 0.f;
        for (int s = sbeg; s < sbeg + 256; ++s) {
            const unsigned int u = eb[(size_t)s * (Ee >> 1)];
            const float w = s_sc[s];
            a0 = fmaf(w, bflo(u), a0);
            a1 = fmaf(w, bfhi(u), a1);
        }
        const int e2 = (tid & 255) * 2;
        s_ctx[tid >> 8][e2]     = a0;
        s_ctx[tid >> 8][e2 + 1] = a1;
    }
    __syncthreads();
    const float cv = s_ctx[0][tid] + s_ctx[1][tid];
    ctx[(size_t)b * Ee + tid]    = cv;
    ctx_bf[(size_t)b * Ee + tid] = f2bf(cv);
}

// ---- LSTM gates (i,f,g,o) + state update; also emits h in bf16 ----
__global__ __launch_bounds__(256) void lstm_gate_kernel(
    const float* __restrict__ z, float* __restrict__ h,
    __bf16* __restrict__ h_bf, float* __restrict__ c)
{
    const int idx = blockIdx.x * blockDim.x + threadIdx.x;
    const int b = idx >> 9;
    const int u = idx & (Uu - 1);
    const float* zr = z + (size_t)b * U4;
    const float zi = zr[u];
    const float zf = zr[Uu + u];
    const float zg = zr[2 * Uu + u];
    const float zo = zr[3 * Uu + u];
    const float si = 1.f / (1.f + __expf(-zi));
    const float sf = 1.f / (1.f + __expf(-zf));
    const float so = 1.f / (1.f + __expf(-zo));
    const float cn = sf * c[idx] + si * tanhf(zg);
    const float hn = so * tanhf(cn);
    c[idx] = cn;
    h[idx] = hn;
    h_bf[idx] = f2bf(hn);
}

extern "C" void kernel_launch(void* const* d_in, const int* in_sizes, int n_in,
                              void* d_out, int out_size, void* d_ws, size_t ws_size,
                              hipStream_t stream) {
    (void)in_sizes; (void)n_in; (void)out_size; (void)ws_size;
    const float* h0  = (const float*)d_in[0];
    const float* c0  = (const float*)d_in[1];
    const float* enc = (const float*)d_in[2];
    const float* W1  = (const float*)d_in[3];
    const float* b1  = (const float*)d_in[4];
    const float* W2  = (const float*)d_in[5];
    const float* b2  = (const float*)d_in[6];
    const float* Vw  = (const float*)d_in[7];
    const float* Vb  = (const float*)d_in[8];
    const float* lk  = (const float*)d_in[9];
    const float* lr  = (const float*)d_in[10];
    const float* lb  = (const float*)d_in[11];
    const float* fw  = (const float*)d_in[12];
    const float* fb  = (const float*)d_in[13];
    float* out = (float*)d_out;

    char* cur = (char*)d_ws;
    auto take = [&](size_t bytes) { char* p = cur; cur += (bytes + 255) & ~(size_t)255; return p; };
    __bf16* enc_bf = (__bf16*)take((size_t)Bb * Ss * Ee * 2);
    __bf16* ep_bf  = (__bf16*)take((size_t)Bb * Ss * Uu * 2);
    __bf16* W1t    = (__bf16*)take((size_t)Uu * Uu * 2);
    __bf16* W2t    = (__bf16*)take((size_t)Ee * Uu * 2);
    __bf16* lkT    = (__bf16*)take((size_t)Tt * U4 * Ee * 2);
    __bf16* lrT    = (__bf16*)take((size_t)Tt * U4 * Uu * 2);
    __bf16* fwT    = (__bf16*)take((size_t)Tt * Oo * Uu * 2);
    float*  q      = (float*)take((size_t)Bb * Uu * 4);
    float*  ctx    = (float*)take((size_t)Bb * Ee * 4);
    __bf16* ctx_bf = (__bf16*)take((size_t)Bb * Ee * 2);
    float*  z      = (float*)take((size_t)Bb * U4 * 4);
    float*  h      = (float*)take((size_t)Bb * Uu * 4);
    float*  c      = (float*)take((size_t)Bb * Uu * 4);
    __bf16* h_bf   = (__bf16*)take((size_t)Bb * Uu * 2);

    {
        int n4 = (Bb * Ss * Ee) / 4;
        cvt_bf16_kernel<<<n4 / 256, 256, 0, stream>>>(enc, enc_bf, n4);
    }
    transpose_bf16_kernel<<<dim3(Uu / 32, Uu / 32, 1),  256, 0, stream>>>(W1, W1t, Uu, Uu);
    transpose_bf16_kernel<<<dim3(Uu / 32, Ee / 32, 1),  256, 0, stream>>>(W2, W2t, Ee, Uu);
    transpose_bf16_kernel<<<dim3(U4 / 32, Ee / 32, Tt), 256, 0, stream>>>(lk, lkT, Ee, U4);
    transpose_bf16_kernel<<<dim3(U4 / 32, Uu / 32, Tt), 256, 0, stream>>>(lr, lrT, Uu, U4);
    transpose_bf16_kernel<<<dim3(Oo / 32, Uu / 32, Tt), 256, 0, stream>>>(fw, fwT, Uu, Oo);
    {
        int n4 = (Bb * Uu) / 4;
        cvt_bf16_kernel<<<n4 / 256, 256, 0, stream>>>(h0, h_bf, n4);
    }
    (void)hipMemcpyAsync(h, h0, sizeof(float) * Bb * Uu, hipMemcpyDeviceToDevice, stream);
    (void)hipMemcpyAsync(c, c0, sizeof(float) * Bb * Uu, hipMemcpyDeviceToDevice, stream);

    auto gemm = [&](const __bf16* A, const __bf16* Bt, const float* bias,
                    float* Cf, __bf16* Cb, int M, int N, int K, int ldc,
                    int accum, int outBf) {
        int blocks = (M / 128) * (N / 64);
        gemm_bf16_wmma<<<blocks, 256, 0, stream>>>(A, Bt, bias, Cf, Cb,
                                                   M, N, K, ldc, accum, outBf);
    };

    // Hoisted: enc_proj = enc_out @ W2 + b2 -> bf16
    gemm(enc_bf, W2t, b2, nullptr, ep_bf, Bb * Ss, Uu, Ee, Uu, 0, 1);

    float* probs = out + (size_t)Bb * Tt * Oo;
    for (int t = 0; t < Tt; ++t) {
        gemm(h_bf, W1t, b1, q, nullptr, Bb, Uu, Uu, Uu, 0, 0);
        attention_kernel<<<Bb, 512, 0, stream>>>(ep_bf, q, Vw, Vb, enc_bf,
                                                 ctx, ctx_bf, probs, t);
        gemm(ctx_bf, lkT + (size_t)t * U4 * Ee, lb + (size_t)t * U4,
             z, nullptr, Bb, U4, Ee, U4, 0, 0);
        gemm(h_bf,   lrT + (size_t)t * U4 * Uu, nullptr,
             z, nullptr, Bb, U4, Uu, U4, 1, 0);
        lstm_gate_kernel<<<(Bb * Uu) / 256, 256, 0, stream>>>(z, h, h_bf, c);
        gemm(h_bf, fwT + (size_t)t * Oo * Uu, fb + (size_t)t * Oo,
             out + (size_t)t * Oo, nullptr, Bb, Oo, Uu, Tt * Oo, 0, 0);
    }
}
